// LSTM_20100446945319
// MI455X (gfx1250) — matrix operand order repaired and checked by reading, not compiled, not run on hardware
//
#include <hip/hip_runtime.h>

// LayerNorm-LSTM, B=512 T=512 D=128 H=256, fp32 in/out, bf16 WMMA compute.
// Persistent kernel: 32 workgroups x 512 threads; each WG owns 16 batch rows
// for all 512 timesteps (recurrence is batch-independent -> no grid sync).
// W-part of packed weights (256KB) lives in LDS; U-part (512KB) streams from L2
// via global_load_b128 (addrspace(1) cast keeps it off the FLAT/DScnt path).

typedef __attribute__((ext_vector_type(16))) __bf16        v16bf;
typedef __attribute__((ext_vector_type(8)))  float         v8f;
typedef __attribute__((ext_vector_type(4)))  unsigned int  v4u;   // POD 128-bit chunk
typedef const v4u __attribute__((address_space(1)))* gq_t;        // global-only b128 loads

#define T_STEPS 512
#define Dd      128
#define Hh      256
#define KT      12            // K tiles of 32 over K=384 = D+H (4 W-tiles + 8 U-tiles)
#define NT      64            // N tiles of 16 over 4H=1024
#define ROW_US  392           // padded LDS A-row stride in ushorts (784 B)
#define W_USH   (NT * 4 * 512)   // 131072 ushorts = 256 KB  (LDS-resident part)
#define LN_EPSF 1e-3f

__device__ __forceinline__ unsigned short f2bf(float f) {
  union { float f; unsigned u; } v; v.f = f;
  unsigned u = v.u;
  return (unsigned short)((u + 0x7FFFu + ((u >> 16) & 1u)) >> 16); // RNE
}

union FragU { v16bf v; v4u q[2]; };

// ---- one-time weight pack: concat([W;U]) f32 -> bf16 B-fragments in d_ws ----
// B-fragment element map (16-bit B 32x16): n = lane&15, k = (p<8) ? 8*(lane>>4)+p : 16+8*(lane>>4)+(p-8).
// kt<4 (W part): chunk-split layout for LDS (q0 plane of all lanes, then q1).
// kt>=4 (U part): per-lane-contiguous 32B for coalesced global_load_b128 pairs.
__global__ void prep_wu(const float* __restrict__ Wi, const float* __restrict__ Wf,
                        const float* __restrict__ Wc, const float* __restrict__ Wo,
                        const float* __restrict__ Ui, const float* __restrict__ Uf,
                        const float* __restrict__ Uc, const float* __restrict__ Uo,
                        unsigned short* __restrict__ wu) {
  int idx = blockIdx.x * blockDim.x + threadIdx.x;     // ((nt*12+kt)*32+lane)*16+p
  if (idx >= NT * KT * 32 * 16) return;
  int p    = idx & 15;
  int lane = (idx >> 4) & 31;
  int r2   = idx >> 9;
  int kt   = r2 % KT;
  int nt   = r2 / KT;
  int k    = kt * 32 + ((p < 8) ? (8 * (lane >> 4) + p) : (16 + 8 * (lane >> 4) + (p - 8)));
  int nglob = nt * 16 + (lane & 15);
  int g  = nglob >> 8;
  int jg = nglob & 255;
  const float* Wg[4] = { Wi, Wf, Wc, Wo };
  const float* Ug[4] = { Ui, Uf, Uc, Uo };
  float v = (k < Dd) ? Wg[g][k * Hh + jg] : Ug[g][(k - Dd) * Hh + jg];

  int dst;
  if (kt < 4) {
    int base = (nt * 4 + kt) * 512;
    dst = base + ((p < 8) ? (lane * 8 + p) : (256 + lane * 8 + (p - 8)));
  } else {
    dst = W_USH + ((nt * 8 + (kt - 4)) * 32 + lane) * 16 + p;
  }
  wu[dst] = f2bf(v);
}

// ---- persistent recurrent kernel ----
__global__ __launch_bounds__(512, 1)
void lstm_persist(const float* __restrict__ x,
                  const unsigned short* __restrict__ wu,
                  const float* __restrict__ bi, const float* __restrict__ bff,
                  const float* __restrict__ bc, const float* __restrict__ bo,
                  const float* __restrict__ gamma, const float* __restrict__ beta,
                  float* __restrict__ out) {
  __shared__ unsigned short As[16 * ROW_US];     // A = [x_t (K 0..127) | h (K 128..383)]
  __shared__ float rsum[2][16], rsumsq[2][16];   // double-buffered LN reductions
  extern __shared__ unsigned short dynW[];       // 131072 ushorts = 256 KB W-fragments

  const int tid  = threadIdx.x;
  const int wave = tid >> 5;
  const int lane = tid & 31;
  const int hi   = lane >> 4;
  const int n    = lane & 15;
  const int j    = wave * 16 + n;                // hidden column owned by this lane
  const int batch0 = blockIdx.x * 16;

  const float bias0 = bi[j], bias1 = bff[j], bias2 = bc[j], bias3 = bo[j];
  const float gam = gamma[j], bet = beta[j];

  // stage W-part fragments into LDS (linear 256 KB copy, b128 granularity)
  {
    const v4u* src = (const v4u*)wu;
    v4u* dst = (v4u*)dynW;
    for (int i = tid; i < W_USH / 8; i += 512) dst[i] = src[i];
  }

  // zero the h region of A (cols 128..383); x region fully overwritten each step
  for (int i = tid; i < 16 * 256; i += 512) {
    int row = i >> 8, col = 128 + (i & 255);
    As[row * ROW_US + col] = 0;
  }

  float c_reg[8], hn[8];
#pragma unroll
  for (int r = 0; r < 8; ++r) { c_reg[r] = 0.f; hn[r] = 0.f; }

  // A-fragment: row = n; chunk0 at kt*64 + hi*16 bytes, chunk1 at +32
  const char* Ab  = (const char*)As;
  const int   arow = n * (ROW_US * 2);

  // x-tile staging: thread -> (row xb, 4 contiguous d)
  const int xb = tid >> 5;
  const int xd = (tid & 31) * 4;
  const float* xrow = x + ((size_t)(batch0 + xb) * T_STEPS) * Dd + xd;

  const unsigned short* wuUbase = wu + W_USH;    // U-part base (kernel-arg derived)
  unsigned int uoff = 0;                         // anti-hoist offset (stays 0)

  for (int t = 0; t < T_STEPS; ++t) {
    // phase 1: stage x_t (f32 -> bf16) into LDS, zero this step's LN buffers
    float4 xv = *(const float4*)(xrow + (size_t)t * Dd);
    uint2 pk;
    pk.x = (unsigned)f2bf(xv.x) | ((unsigned)f2bf(xv.y) << 16);
    pk.y = (unsigned)f2bf(xv.z) | ((unsigned)f2bf(xv.w) << 16);
    *(uint2*)((char*)As + xb * (ROW_US * 2) + xd * 2) = pk;
    if (tid < 16) { rsum[t & 1][tid] = 0.f; rsumsq[t & 1][tid] = 0.f; }
    __syncthreads();

    // anti-hoist: redefine the (always-zero) offset each iteration so LLVM
    // cannot CSE the weight loads across timesteps (prevents hoist->spill),
    // while the pointer itself stays kernel-arg derived -> global_load path.
    asm volatile("" : "+s"(uoff));

    // phase 2: g = [x_t|h] @ [W;U] + b   (wave w owns gate slices of cols j)
    v8f acc[4];
#pragma unroll
    for (int g = 0; g < 4; ++g) {
      float bv = (g == 0) ? bias0 : (g == 1) ? bias1 : (g == 2) ? bias2 : bias3;
      acc[g] = (v8f){ bv, bv, bv, bv, bv, bv, bv, bv };
    }
#pragma unroll
    for (int kt = 0; kt < KT; ++kt) {
      FragU fa;
      const char* abase = Ab + arow + kt * 64 + hi * 16;
      fa.q[0] = *(const v4u*)(abase);
      fa.q[1] = *(const v4u*)(abase + 32);
#pragma unroll
      for (int g = 0; g < 4; ++g) {
        int nt = g * 16 + wave;
        FragU fb;
        if (kt < 4) {
          const v4u* wp = (const v4u*)(dynW + (nt * 4 + kt) * 512);
          fb.q[0] = wp[lane];        // chunk0 plane
          fb.q[1] = wp[32 + lane];   // chunk1 plane (+512B)
        } else {
          gq_t up = (gq_t)(wuUbase + ((size_t)((nt * 8 + (kt - 4)) * 32 + lane) * 16 + uoff));
          fb.q[0] = up[0];
          fb.q[1] = up[1];
        }
        acc[g] = __builtin_amdgcn_wmma_f32_16x16x32_bf16(
            false, fa.v, false, fb.v, (short)0, acc[g], false, false);
      }
    }

    // phase 3: gates + cell update (row b = r + 8*hi, col j)
    float hr[8];
#pragma unroll
    for (int r = 0; r < 8; ++r) {
      float iv = 1.f / (1.f + __expf(-acc[0][r]));
      float fv = 1.f / (1.f + __expf(-acc[1][r]));
      float cv = 1.f - 2.f / (__expf(2.f * acc[2][r]) + 1.f);   // tanh
      float ov = 1.f / (1.f + __expf(-acc[3][r]));
      float c  = fv * c_reg[r] + iv * cv;
      c_reg[r] = c;
      hr[r] = ov * (1.f - 2.f / (__expf(2.f * c) + 1.f));
    }

    // phase 4: LN row reductions (shfl tree over the 16-lane half, then LDS atomics)
#pragma unroll
    for (int r = 0; r < 8; ++r) {
      float s = hr[r], q = hr[r] * hr[r];
      for (int m = 1; m <= 8; m <<= 1) {
        s += __shfl_xor(s, m, 32);
        q += __shfl_xor(q, m, 32);
      }
      if (n == 0) {
        int b = r + 8 * hi;
        atomicAdd(&rsum[t & 1][b], s);
        atomicAdd(&rsumsq[t & 1][b], q);
      }
    }
    __syncthreads();

    // phase 5: normalize and stage h_ln back into A for the next step
#pragma unroll
    for (int r = 0; r < 8; ++r) {
      int b = r + 8 * hi;
      float mu  = rsum[t & 1][b] * (1.f / 256.f);
      float var = rsumsq[t & 1][b] * (1.f / 256.f) - mu * mu;
      float v = gam * (hr[r] - mu) * __frsqrt_rn(var + LN_EPSF) + bet;
      hn[r] = v;
      *((unsigned short*)((char*)As + b * (ROW_US * 2) + (Dd + j) * 2)) = f2bf(v);
    }
    // next iteration's first barrier orders these h writes vs. A reads
  }

#pragma unroll
  for (int r = 0; r < 8; ++r) {
    int b = r + 8 * hi;
    out[(size_t)(batch0 + b) * Hh + j] = hn[r];
  }
}

extern "C" void kernel_launch(void* const* d_in, const int* in_sizes, int n_in,
                              void* d_out, int out_size, void* d_ws, size_t ws_size,
                              hipStream_t stream) {
  const float* x  = (const float*)d_in[0];
  const float* Wi = (const float*)d_in[1];
  const float* Wf = (const float*)d_in[2];
  const float* Wc = (const float*)d_in[3];
  const float* Wo = (const float*)d_in[4];
  const float* Ui = (const float*)d_in[5];
  const float* Uf = (const float*)d_in[6];
  const float* Uc = (const float*)d_in[7];
  const float* Uo = (const float*)d_in[8];
  const float* bi = (const float*)d_in[9];
  const float* bf = (const float*)d_in[10];
  const float* bc = (const float*)d_in[11];
  const float* bo = (const float*)d_in[12];
  const float* gm = (const float*)d_in[13];
  const float* bt = (const float*)d_in[14];
  unsigned short* wu = (unsigned short*)d_ws;      // 768 KB packed bf16 fragments

  int total = NT * KT * 32 * 16;
  prep_wu<<<(total + 255) / 256, 256, 0, stream>>>(Wi, Wf, Wc, Wo, Ui, Uf, Uc, Uo, wu);
  lstm_persist<<<32, 512, W_USH * sizeof(unsigned short), stream>>>(
      x, wu, bi, bf, bc, bo, gm, bt, (float*)d_out);
}
